// LSTM_9964324126771
// MI455X (gfx1250) — matrix operand order repaired and checked by reading, not compiled, not run on hardware
//
#include <hip/hip_runtime.h>

// ---------------------------------------------------------------------------
// LSTM  B=256, S=512, I=64, H=256  — CDNA5 (gfx1250, wave32) WMMA bf16
//
// Phase 0: transpose+convert w_g [64,256], u_g [256,256] fp32 -> bf16
//          column-major; pack biases.
// Phase 1: pre[g][s][btile][nt][lane][8] (C-fragment order, fp32) =
//          x @ w_g + b_g  via v_wmma_f32_16x16x32_bf16 (K=64 -> 2 WMMAs).
// Phase 2: persistent recurrence. 16 blocks (16-row batch slice each),
//          16 waves/block (16-col N-tile each). u fragments register-
//          resident (4 gates x 8 kblocks x v16bf), h via LDS bf16,
//          c in registers, vectorized v8f pre loads, 2 barriers/step.
// ---------------------------------------------------------------------------

typedef __attribute__((ext_vector_type(16))) __bf16 v16bf;
typedef __attribute__((ext_vector_type(8)))  float  v8f;

#define GATE_STRIDE 33554432   // 512*16*16*256  (elements per gate in `pre`)

static __device__ inline unsigned short f2bf(float f) {
    union { float f; unsigned u; } v; v.f = f;
    unsigned r = v.u + 0x7FFFu + ((v.u >> 16) & 1u);   // round-to-nearest-even
    return (unsigned short)(r >> 16);
}

static __device__ inline float sigm(float x) {
    return 1.0f / (1.0f + __expf(-x));
}
static __device__ inline float tanh_fast(float x) {
    float t = __expf(-2.0f * fabsf(x));
    float r = (1.0f - t) / (1.0f + t);
    return x < 0.0f ? -r : r;
}

static __device__ inline v8f wmma_bf16(v16bf a, v16bf b, v8f c) {
    return __builtin_amdgcn_wmma_f32_16x16x32_bf16(
        /*neg_a=*/false, a, /*neg_b=*/false, b,
        /*c_mod=*/(short)0, c, /*reuse_a=*/false, /*reuse_b=*/false);
}

// A-matrix 16x32 bf16 fragment (ISA 7.12.2): lane m = lane&15.
// lanes 0-15 : V0..3 -> K 0..7,  V4..7 -> K 16..23
// lanes 16-31: V0..3 -> K 8..15, V4..7 -> K 24..31
static __device__ inline v16bf load_afrag(const unsigned short* src, int m,
                                          int kb, int hi, int stride) {
    union { v16bf v; unsigned u[8]; } r;
#pragma unroll
    for (int v8 = 0; v8 < 8; ++v8) {
        int kp = (v8 < 4) ? ((hi ? 8 : 0) + 2 * v8)
                          : ((hi ? 24 : 16) + 2 * (v8 - 4));
        r.u[v8] = *(const unsigned*)(src + m * stride + kb * 32 + kp);
    }
    return r.v;
}

// B-matrix 32x16 bf16 fragment: lane n = lane&15.
// lanes 0-15: V0..3 -> K 0..7, V4..7 -> K 16..23 ; lanes 16-31: V0..3 -> K 8..15, V4..7 -> K 24..31
// src is column-major [n][k] with row length `stride`.
static __device__ inline v16bf load_bfrag(const unsigned short* src, int n,
                                          int kb, int hi, int stride) {
    union { v16bf v; unsigned u[8]; } r;
#pragma unroll
    for (int v8 = 0; v8 < 8; ++v8) {
        int kp = (v8 < 4) ? ((hi ? 8 : 0) + 2 * v8)
                          : ((hi ? 24 : 16) + 2 * (v8 - 4));
        r.u[v8] = *(const unsigned*)(src + n * stride + kb * 32 + kp);
    }
    return r.v;
}

// --------------------------- Phase 0: convert --------------------------------
// out[n*K + k] = bf16(in[k*N + n])
__global__ __launch_bounds__(256) void cvt_transpose(
    const float* __restrict__ in, unsigned short* __restrict__ outp,
    int K, int N) {
    int idx = blockIdx.x * blockDim.x + threadIdx.x;
    if (idx >= K * N) return;
    int k = idx % K;
    int n = idx / K;
    outp[idx] = f2bf(in[k * N + n]);
}

__global__ __launch_bounds__(256) void pack_bias(
    const float* __restrict__ b0, const float* __restrict__ b1,
    const float* __restrict__ b2, const float* __restrict__ b3,
    float* __restrict__ outp) {
    int i = blockIdx.x * 256 + threadIdx.x;   // 0..1023
    const float* p = (i < 256) ? b0 : (i < 512) ? b1 : (i < 768) ? b2 : b3;
    outp[i] = p[i & 255];
}

// --------------------------- Phase 1: x projections --------------------------
// grid: 8192 blocks = S(512) * B/16(16). block: 256 threads = 8 waves.
__global__ __launch_bounds__(256) void lstm_xproj(
    const float* __restrict__ x,            // [B][S][64] fp32
    const unsigned short* __restrict__ wT,  // [4][256 n][64 k] bf16
    const float* __restrict__ bias,         // [4][256]
    float* __restrict__ pre) {              // [4][S][16][16][32][8] fp32
    __shared__ unsigned short xs[16 * 64];

    int bx    = blockIdx.x;
    int s     = bx >> 4;
    int btile = bx & 15;
    int tid   = threadIdx.x;

    // stage x tile [16 rows][64 k] -> bf16 LDS
    {
        int m  = tid >> 4;            // 0..15
        int k0 = (tid & 15) * 4;      // 0..60
        int b  = btile * 16 + m;
        const float4 f = *(const float4*)(x + ((size_t)b * 512 + s) * 64 + k0);
        xs[m * 64 + k0 + 0] = f2bf(f.x);
        xs[m * 64 + k0 + 1] = f2bf(f.y);
        xs[m * 64 + k0 + 2] = f2bf(f.z);
        xs[m * 64 + k0 + 3] = f2bf(f.w);
    }
    __syncthreads();

    int lane = tid & 31, w = tid >> 5;
    int hi = lane >> 4, mA = lane & 15;

    v16bf a0 = load_afrag(xs, mA, 0, hi, 64);
    v16bf a1 = load_afrag(xs, mA, 1, hi, 64);

#pragma unroll
    for (int p = 0; p < 8; ++p) {
        int id = p * 8 + w;           // 0..63 across the 8 waves
        int g  = id >> 4;             // gate 0..3 (i,f,c,o)
        int nt = id & 15;             // N-tile
        int n  = nt * 16 + (lane & 15);

        const unsigned short* wb = wT + g * 16384;   // [256][64]
        v16bf b0 = load_bfrag(wb, n, 0, hi, 64);
        v16bf b1 = load_bfrag(wb, n, 1, hi, 64);

        v8f acc = {};
        acc = wmma_bf16(a0, b0, acc);
        acc = wmma_bf16(a1, b1, acc);

        float bv = bias[g * 256 + n];
        v8f outv;
#pragma unroll
        for (int v = 0; v < 8; ++v) outv[v] = acc[v] + bv;

        // fragment-ordered store: one contiguous 32B chunk per lane
        int off = ((((g * 512 + s) * 16 + btile) * 16 + nt) * 32 + lane) * 8;
        *(v8f*)(pre + off) = outv;
    }
}

// --------------------------- Phase 2: recurrence -----------------------------
// grid: 16 blocks (batch slices). block: 512 threads = 16 waves (N-tiles).
__global__ __launch_bounds__(512) void lstm_recur(
    const float* __restrict__ pre,          // [4][512][16][16][32][8]
    const unsigned short* __restrict__ uT,  // [4][256 n][256 k] bf16
    float* __restrict__ out,                // [512][256][256]
    float* __restrict__ ht,                 // [256][256]
    float* __restrict__ ct) {               // [256][256]
    __shared__ unsigned short hsh[16 * 256];   // h slice, bf16 row-major

    int tid = threadIdx.x, lane = tid & 31, w = tid >> 5;
    int hi = lane >> 4, mA = lane & 15;
    int btile = blockIdx.x;
    int bBase = btile * 16;
    int nglob = w * 16 + (lane & 15);

#pragma unroll
    for (int j = 0; j < 8; ++j) hsh[tid * 8 + j] = 0;   // h0 = 0

    // Preload this wave's u fragments into registers: 4 gates x 8 kblocks.
    v16bf ub[4][8];
#pragma unroll
    for (int g = 0; g < 4; ++g)
#pragma unroll
        for (int kb = 0; kb < 8; ++kb)
            ub[g][kb] = load_bfrag(uT + g * 65536, nglob, kb, hi, 256);

    float cc[8];
#pragma unroll
    for (int v = 0; v < 8; ++v) cc[v] = 0.0f;

    // per-(gate) fragment-ordered pre base for this (btile, wave, lane)
    int pfrag = ((btile * 16 + w) * 32 + lane) * 8;

    __syncthreads();

    for (int s = 0; s < 512; ++s) {
        // h @ u for the 4 gates (A from LDS, B resident in VGPRs)
        v16bf af[8];
#pragma unroll
        for (int kb = 0; kb < 8; ++kb)
            af[kb] = load_afrag(hsh, mA, kb, hi, 256);

        // vectorized pre-activation loads (overlap with WMMA issue)
        v8f pg[4];
#pragma unroll
        for (int g = 0; g < 4; ++g)
            pg[g] = *(const v8f*)(pre + g * GATE_STRIDE + s * 65536 + pfrag);

        v8f acc[4];
#pragma unroll
        for (int g = 0; g < 4; ++g) acc[g] = (v8f){};
        // kb outer / g inner: consecutive WMMAs hit independent accumulators
#pragma unroll
        for (int kb = 0; kb < 8; ++kb)
#pragma unroll
            for (int g = 0; g < 4; ++g)
                acc[g] = wmma_bf16(af[kb], ub[g][kb], acc[g]);

        __syncthreads();   // all waves done reading hsh

        if (s + 1 < 512) {  // warm GL2 for next step's pre-activations
            const float* nxt = pre + (s + 1) * 65536 + pfrag;
            __builtin_prefetch(nxt + 0 * GATE_STRIDE, 0, 0);
            __builtin_prefetch(nxt + 1 * GATE_STRIDE, 0, 0);
            __builtin_prefetch(nxt + 2 * GATE_STRIDE, 0, 0);
            __builtin_prefetch(nxt + 3 * GATE_STRIDE, 0, 0);
        }

#pragma unroll
        for (int v = 0; v < 8; ++v) {
            int mrow  = hi ? v + 8 : v;
            int bglob = bBase + mrow;

            float gi = acc[0][v] + pg[0][v];
            float gf = acc[1][v] + pg[1][v];
            float gg = acc[2][v] + pg[2][v];
            float go = acc[3][v] + pg[3][v];

            float it = sigm(gi);
            float ft = sigm(gf);
            float gt = tanh_fast(gg);
            float ot = sigm(go);

            float c = ft * cc[v] + it * gt;
            cc[v] = c;
            float h = ot * tanh_fast(c);

            out[s * 65536 + bglob * 256 + nglob] = h;
            hsh[mrow * 256 + nglob] = f2bf(h);
            if (s == 511) {
                ht[bglob * 256 + nglob] = h;
                ct[bglob * 256 + nglob] = c;
            }
        }
        __syncthreads();   // new h visible before next step's reads
    }
}

// ---------------------------------------------------------------------------
extern "C" void kernel_launch(void* const* d_in, const int* in_sizes, int n_in,
                              void* d_out, int out_size, void* d_ws,
                              size_t ws_size, hipStream_t stream) {
    (void)in_sizes; (void)n_in; (void)out_size; (void)ws_size;

    const float* x = (const float*)d_in[0];
    const float* W[4] = {(const float*)d_in[1], (const float*)d_in[4],
                         (const float*)d_in[7], (const float*)d_in[10]};
    const float* U[4] = {(const float*)d_in[2], (const float*)d_in[5],
                         (const float*)d_in[8], (const float*)d_in[11]};
    const float* Bv[4] = {(const float*)d_in[3], (const float*)d_in[6],
                          (const float*)d_in[9], (const float*)d_in[12]};

    // ws: wT (4*256*64 bf16) | uT (4*256*256 bf16) | bias (4*256 f32)
    //     | pre (4*512*256*256 f32, 32B-aligned)
    unsigned short* wT = (unsigned short*)d_ws;
    unsigned short* uT = wT + 4 * 256 * 64;
    float* bias = (float*)(uT + 4 * 256 * 256);
    float* pre  = bias + 4 * 256;

    for (int g = 0; g < 4; ++g)
        cvt_transpose<<<(16384 + 255) / 256, 256, 0, stream>>>(
            W[g], wT + g * 16384, 64, 256);
    for (int g = 0; g < 4; ++g)
        cvt_transpose<<<(65536 + 255) / 256, 256, 0, stream>>>(
            U[g], uT + g * 65536, 256, 256);
    pack_bias<<<4, 256, 0, stream>>>(Bv[0], Bv[1], Bv[2], Bv[3], bias);

    lstm_xproj<<<8192, 256, 0, stream>>>(x, wT, bias, pre);

    float* out = (float*)d_out;
    float* ht  = out + 512 * 256 * 256;
    float* ct  = ht + 256 * 256;
    lstm_recur<<<16, 512, 0, stream>>>(pre, uT, out, ht, ct);
}
